// GPT_57543971832749
// MI455X (gfx1250) — matrix operand order repaired and checked by reading, not compiled, not run on hardware
//
#include <hip/hip_runtime.h>
#include <hip/hip_bf16.h>

// ---------------------------------------------------------------------------
// Model constants (must match reference)
// ---------------------------------------------------------------------------
#define BB      8
#define GRID    16
#define DMODEL  1024
#define NH      16
#define NKVH    8
#define NL      8
#define HD      64
#define SEQ     (1 + GRID * GRID)        // 257
#define MROWS   (BB * SEQ)               // 2056
#define TKV     ((NH + 2 * NKVH) * HD)   // 2048
#define HIDDEN  2816
#define VOCAB   16384
#define EPSF    1e-5f

typedef unsigned short u16;
typedef unsigned int   u32;

typedef __bf16 v16bf __attribute__((ext_vector_type(16)));
typedef float  v8f   __attribute__((ext_vector_type(8)));

union FragBF { uint4 q[2]; v16bf v; };

static __device__ __forceinline__ u16 f2bf(float f) {
    u32 u = __float_as_uint(f);
    u32 r = u + 0x7FFFu + ((u >> 16) & 1u);   // round-to-nearest-even
    return (u16)(r >> 16);
}
static __device__ __forceinline__ float bf2f(u16 h) {
    return __uint_as_float(((u32)h) << 16);
}

// ---------------------------------------------------------------------------
// 1) fp32 (K,N) -> bf16 (N,K) tiled transpose (weights pre-pass)
// ---------------------------------------------------------------------------
__global__ __launch_bounds__(256) void k_transpose_bf16(
    const float* __restrict__ W, u16* __restrict__ Out, int K, int N) {
    __shared__ float tile[32][33];
    const size_t lsz = (size_t)K * N;
    W   += (size_t)blockIdx.z * lsz;
    Out += (size_t)blockIdx.z * lsz;
    const int n0 = blockIdx.x * 32, k0 = blockIdx.y * 32;
    const int tx = threadIdx.x, ty = threadIdx.y;
#pragma unroll
    for (int i = 0; i < 4; ++i)
        tile[ty + i * 8][tx] = W[(size_t)(k0 + ty + i * 8) * N + n0 + tx];
    __syncthreads();
#pragma unroll
    for (int i = 0; i < 4; ++i)
        Out[(size_t)(n0 + ty + i * 8) * K + k0 + tx] = f2bf(tile[tx][ty + i * 8]);
}

// ---------------------------------------------------------------------------
// 2) Embedding gather: h[b,s,:] = s==0 ? cls_table[cond] : tok_table[idx]
// ---------------------------------------------------------------------------
__global__ __launch_bounds__(256) void k_embed(
    const int* __restrict__ idx, const int* __restrict__ cond_idx,
    const float* __restrict__ tok_table, const float* __restrict__ cls_table,
    float* __restrict__ h) {
    const int row = blockIdx.x;           // 0..MROWS-1
    const int b = row / SEQ, s = row % SEQ;
    const float* src = (s == 0)
        ? cls_table + (size_t)cond_idx[b] * DMODEL
        : tok_table + (size_t)idx[b * (SEQ - 1) + (s - 1)] * DMODEL;
    ((float4*)(h + (size_t)row * DMODEL))[threadIdx.x] =
        ((const float4*)src)[threadIdx.x];
}

// ---------------------------------------------------------------------------
// 3) RMSNorm (fp32 in, bf16 out), one row (D=1024) per block of 256
// ---------------------------------------------------------------------------
__global__ __launch_bounds__(256) void k_rmsnorm_bf16(
    const float* __restrict__ h, const float* __restrict__ w,
    u16* __restrict__ out) {
    __shared__ float red[256];
    const int row = blockIdx.x, tid = threadIdx.x;
    const float4 v = ((const float4*)(h + (size_t)row * DMODEL))[tid];
    float ss = v.x * v.x + v.y * v.y + v.z * v.z + v.w * v.w;
    red[tid] = ss;
    __syncthreads();
    for (int off = 128; off; off >>= 1) {
        if (tid < off) red[tid] += red[tid + off];
        __syncthreads();
    }
    const float rms = rsqrtf(red[0] * (1.0f / DMODEL) + EPSF);
    const float4 wv = ((const float4*)w)[tid];
    u32 lo = (u32)f2bf(v.x * rms * wv.x) | ((u32)f2bf(v.y * rms * wv.y) << 16);
    u32 hi = (u32)f2bf(v.z * rms * wv.z) | ((u32)f2bf(v.w * rms * wv.w) << 16);
    ((uint2*)(out + (size_t)row * DMODEL))[tid] = make_uint2(lo, hi);
}

// ---------------------------------------------------------------------------
// 4) WMMA bf16 GEMM:  C[M,N] = A[M,K] (bf16 row-major) x W  with W given as
//    Bt[N,K] (bf16, pre-transposed).  mode 0: Out=f32 store; 1: Out f32 +=
//    Macro-tile 128x128, 8 waves (4 M x 2 N), wave tile 32x64, K-step 32.
// ---------------------------------------------------------------------------
#define LDSW 40   // padded row stride (bf16 elems) for bank spread
__global__ __launch_bounds__(256) void k_gemm_bf16(
    const u16* __restrict__ A, const u16* __restrict__ Bt,
    float* __restrict__ Out, int M, int N, int K, int mode) {
    __shared__ u16 As[128 * LDSW];
    __shared__ u16 Bs[128 * LDSW];

    const int tid  = threadIdx.x;
    const int lane = tid & 31, wave = tid >> 5;
    const int wm = wave & 3, wn = wave >> 2;         // 4 x 2 wave grid
    const int bm = blockIdx.y * 128, bn = blockIdx.x * 128;

    const int sr = tid >> 1;        // staging row 0..127
    const int sh = tid & 1;         // staging half (16 bf16 = 32B each)
    const int arow = bm + sr;       // global A row (guarded)
    const int brow = bn + sr;       // global Bt row (N multiple of 128)

    v8f acc[2][4];
#pragma unroll
    for (int i = 0; i < 2; ++i)
#pragma unroll
        for (int j = 0; j < 4; ++j) acc[i][j] = (v8f){0.f,0.f,0.f,0.f,0.f,0.f,0.f,0.f};

    const int l15 = lane & 15, lh = lane >> 4;

    for (int kk = 0; kk < K; kk += 32) {
        __syncthreads();
        // stage A (zero-guard rows past M are never consumed by the epilogue)
        if (arow < M) {
            const uint4* gp = (const uint4*)(A + (size_t)arow * K + kk + sh * 16);
            uint4* sp = (uint4*)&As[sr * LDSW + sh * 16];
            sp[0] = gp[0]; sp[1] = gp[1];
            if (kk + 32 < K)
                __builtin_prefetch((const void*)(A + (size_t)arow * K + kk + 32 + sh * 16), 0, 0);
        }
        {
            const uint4* gp = (const uint4*)(Bt + (size_t)brow * K + kk + sh * 16);
            uint4* sp = (uint4*)&Bs[sr * LDSW + sh * 16];
            sp[0] = gp[0]; sp[1] = gp[1];
            if (kk + 32 < K)
                __builtin_prefetch((const void*)(Bt + (size_t)brow * K + kk + 32 + sh * 16), 0, 0);
        }
        __syncthreads();

        // A fragments: 16x32 bf16; lane holds row m=lane&15,
        // K = lh*8..lh*8+7 (VGPR0-3) and 16+lh*8.. (VGPR4-7)
        FragBF a[2], b[4];
#pragma unroll
        for (int mt = 0; mt < 2; ++mt) {
            const int r = wm * 32 + mt * 16 + l15;
            a[mt].q[0] = *(const uint4*)&As[r * LDSW + lh * 8];
            a[mt].q[1] = *(const uint4*)&As[r * LDSW + 16 + lh * 8];
        }
        // B fragments: 32x16 bf16; lane holds column n=lane&15,
        // K = lh*8..lh*8+7 (VGPR0-3) and 16+lh*8.. (VGPR4-7)
#pragma unroll
        for (int nt = 0; nt < 4; ++nt) {
            const int c = wn * 64 + nt * 16 + l15;
            b[nt].q[0] = *(const uint4*)&Bs[c * LDSW + lh * 8];
            b[nt].q[1] = *(const uint4*)&Bs[c * LDSW + 16 + lh * 8];
        }
#pragma unroll
        for (int mt = 0; mt < 2; ++mt)
#pragma unroll
            for (int nt = 0; nt < 4; ++nt)
                acc[mt][nt] = __builtin_amdgcn_wmma_f32_16x16x32_bf16(
                    false, a[mt].v, false, b[nt].v,
                    (short)0, acc[mt][nt], false, false);
    }

    // epilogue: C layout — VGPR j: M = j + 8*(lane>>4), N = lane&15
#pragma unroll
    for (int mt = 0; mt < 2; ++mt)
#pragma unroll
        for (int nt = 0; nt < 4; ++nt) {
            const int gc = bn + wn * 64 + nt * 16 + l15;
#pragma unroll
            for (int j = 0; j < 8; ++j) {
                const int gr = bm + wm * 32 + mt * 16 + lh * 8 + j;
                if (gr < M) {
                    float* p = Out + (size_t)gr * N + gc;
                    if (mode == 1) *p += acc[mt][nt][j];
                    else           *p  = acc[mt][nt][j];
                }
            }
        }
}

// ---------------------------------------------------------------------------
// 5) Attention: one block per (head, batch).  2D RoPE applied in-kernel,
//    keys rope'd into LDS (bf16), flash-style online softmax, V from L2.
// ---------------------------------------------------------------------------
__device__ __forceinline__ void rope_angle(int s, int i, float* c, float* sn) {
    if (s == 0) { *c = 0.f; *sn = 0.f; return; }     // cls token: zero freqs_cis
    const int pos = s - 1, py = pos >> 4, px = pos & 15;
    const int j = (i < 16) ? i : (i - 16);
    const float coord = (i < 16) ? (float)py : (float)px;
    const float ang = coord * __powf(10000.f, -(float)(2 * j) * (1.f / 32.f));
    *c = __cosf(ang); *sn = __sinf(ang);
}

__global__ __launch_bounds__(256, 1) void k_attn(
    const float* __restrict__ qkv, u16* __restrict__ outb) {
    __shared__ u16 Ksh[SEQ * HD];                     // rope'd keys, bf16
    const int hh = blockIdx.x, b = blockIdx.y, tid = threadIdx.x;
    const int kvh = hh >> 1;                         // GQA: rep = NH/NKVH = 2
    const float scale = 0.125f;                      // 1/sqrt(64)

    // phase 1: rope keys into LDS
    for (int s = tid; s < SEQ; s += 256) {
        const float* kr = qkv + (size_t)(b * SEQ + s) * TKV + NH * HD + kvh * HD;
#pragma unroll
        for (int i = 0; i < 32; ++i) {
            float c, sn; rope_angle(s, i, &c, &sn);
            const float x0 = kr[2 * i], x1 = kr[2 * i + 1];
            Ksh[s * HD + 2 * i]     = f2bf(x0 * c - x1 * sn);
            Ksh[s * HD + 2 * i + 1] = f2bf(x1 * c + x0 * sn);
        }
    }
    __syncthreads();

    // phase 2: each thread owns query rows tid, tid+256, ...
    for (int q = tid; q < SEQ; q += 256) {
        float qv[HD];
        const float* qr = qkv + (size_t)(b * SEQ + q) * TKV + hh * HD;
#pragma unroll
        for (int i = 0; i < 32; ++i) {
            float c, sn; rope_angle(q, i, &c, &sn);
            const float x0 = qr[2 * i], x1 = qr[2 * i + 1];
            qv[2 * i]     = x0 * c - x1 * sn;
            qv[2 * i + 1] = x1 * c + x0 * sn;
        }
        float m = -3.0e38f, l = 0.f, o[HD];
#pragma unroll
        for (int d = 0; d < HD; ++d) o[d] = 0.f;
        for (int k = 0; k <= q; ++k) {               // causal
            float sc = 0.f;
#pragma unroll
            for (int d = 0; d < HD; ++d) sc += qv[d] * bf2f(Ksh[k * HD + d]);
            sc *= scale;
            const float nm = fmaxf(m, sc);
            const float corr = __expf(m - nm);
            const float p = __expf(sc - nm);
            const float* vr = qkv + (size_t)(b * SEQ + k) * TKV
                            + (NH + NKVH) * HD + kvh * HD;
            l = l * corr + p;
#pragma unroll
            for (int d = 0; d < HD; ++d) o[d] = o[d] * corr + p * vr[d];
            m = nm;
        }
        u16* op = outb + (size_t)(b * SEQ + q) * DMODEL + hh * HD;
        const float inv = 1.f / l;
#pragma unroll
        for (int d = 0; d < HD; ++d) op[d] = f2bf(o[d] * inv);
    }
}

// ---------------------------------------------------------------------------
// 6) silu(g1) * g3 -> bf16
// ---------------------------------------------------------------------------
__global__ __launch_bounds__(256) void k_silu_mul(
    const float* __restrict__ g1, const float* __restrict__ g3,
    u16* __restrict__ out, size_t n) {
    const size_t i = (size_t)blockIdx.x * 256 + threadIdx.x;
    if (i < n) {
        const float x = g1[i];
        out[i] = f2bf(x * __frcp_rn(1.f + __expf(-x)) * g3[i]);
    }
}

// ---------------------------------------------------------------------------
// Host orchestration
// ---------------------------------------------------------------------------
extern "C" void kernel_launch(void* const* d_in, const int* in_sizes, int n_in,
                              void* d_out, int out_size, void* d_ws, size_t ws_size,
                              hipStream_t stream) {
    (void)in_sizes; (void)n_in; (void)out_size; (void)ws_size;
    const int*   idx      = (const int*)  d_in[0];
    const int*   cond_idx = (const int*)  d_in[1];
    const float* tok_tab  = (const float*)d_in[2];
    const float* cls_tab  = (const float*)d_in[3];
    const float* wqkv     = (const float*)d_in[4];
    const float* wo       = (const float*)d_in[5];
    const float* w1       = (const float*)d_in[6];
    const float* w2       = (const float*)d_in[7];
    const float* w3       = (const float*)d_in[8];
    const float* an_w     = (const float*)d_in[9];
    const float* fn_w     = (const float*)d_in[10];
    const float* final_w  = (const float*)d_in[11];
    const float* head_w   = (const float*)d_in[12];
    float* out = (float*)d_out;

    // workspace carve-out (256B aligned)
    size_t off = 0;
    char* base = (char*)d_ws;
    auto alloc = [&](size_t bytes) -> void* {
        void* p = base + off;
        off += (bytes + 255) & ~(size_t)255;
        return p;
    };
    u16*   wqkvT = (u16*)  alloc((size_t)NL * TKV   * DMODEL * 2);
    u16*   woT   = (u16*)  alloc((size_t)NL * DMODEL* DMODEL * 2);
    u16*   w1T   = (u16*)  alloc((size_t)NL * HIDDEN* DMODEL * 2);
    u16*   w2T   = (u16*)  alloc((size_t)NL * DMODEL* HIDDEN * 2);
    u16*   w3T   = (u16*)  alloc((size_t)NL * HIDDEN* DMODEL * 2);
    u16*   headT = (u16*)  alloc((size_t)VOCAB * DMODEL * 2);
    float* h     = (float*)alloc((size_t)MROWS * DMODEL * 4);
    u16*   xb    = (u16*)  alloc((size_t)MROWS * DMODEL * 2);
    float* qkv   = (float*)alloc((size_t)MROWS * TKV * 4);
    u16*   attnb = (u16*)  alloc((size_t)MROWS * DMODEL * 2);
    float* g1    = (float*)alloc((size_t)MROWS * HIDDEN * 4);
    float* g3    = (float*)alloc((size_t)MROWS * HIDDEN * 4);
    u16*   actb  = (u16*)  alloc((size_t)MROWS * HIDDEN * 2);

    const dim3 tb(32, 8);
    // weight convert + transpose: (K,N) f32 -> (N,K) bf16
    k_transpose_bf16<<<dim3(TKV / 32,   DMODEL / 32, NL), tb, 0, stream>>>(wqkv,  wqkvT, DMODEL, TKV);
    k_transpose_bf16<<<dim3(DMODEL / 32,DMODEL / 32, NL), tb, 0, stream>>>(wo,    woT,   DMODEL, DMODEL);
    k_transpose_bf16<<<dim3(HIDDEN / 32,DMODEL / 32, NL), tb, 0, stream>>>(w1,    w1T,   DMODEL, HIDDEN);
    k_transpose_bf16<<<dim3(DMODEL / 32,HIDDEN / 32, NL), tb, 0, stream>>>(w2,    w2T,   HIDDEN, DMODEL);
    k_transpose_bf16<<<dim3(HIDDEN / 32,DMODEL / 32, NL), tb, 0, stream>>>(w3,    w3T,   DMODEL, HIDDEN);
    k_transpose_bf16<<<dim3(VOCAB / 32, DMODEL / 32, 1),  tb, 0, stream>>>(head_w,headT, DMODEL, VOCAB);

    k_embed<<<MROWS, 256, 0, stream>>>(idx, cond_idx, tok_tab, cls_tab, h);

    const int MB = (MROWS + 127) / 128;   // 17
    for (int l = 0; l < NL; ++l) {
        const u16* lwqkv = wqkvT + (size_t)l * TKV   * DMODEL;
        const u16* lwo   = woT   + (size_t)l * DMODEL* DMODEL;
        const u16* lw1   = w1T   + (size_t)l * HIDDEN* DMODEL;
        const u16* lw2   = w2T   + (size_t)l * DMODEL* HIDDEN;
        const u16* lw3   = w3T   + (size_t)l * HIDDEN* DMODEL;

        k_rmsnorm_bf16<<<MROWS, 256, 0, stream>>>(h, an_w + (size_t)l * DMODEL, xb);
        k_gemm_bf16<<<dim3(TKV / 128, MB), 256, 0, stream>>>(
            xb, lwqkv, qkv, MROWS, TKV, DMODEL, 0);
        k_attn<<<dim3(NH, BB), 256, 0, stream>>>(qkv, attnb);
        k_gemm_bf16<<<dim3(DMODEL / 128, MB), 256, 0, stream>>>(
            attnb, lwo, h, MROWS, DMODEL, DMODEL, 1);

        k_rmsnorm_bf16<<<MROWS, 256, 0, stream>>>(h, fn_w + (size_t)l * DMODEL, xb);
        k_gemm_bf16<<<dim3(HIDDEN / 128, MB), 256, 0, stream>>>(
            xb, lw1, g1, MROWS, HIDDEN, DMODEL, 0);
        k_gemm_bf16<<<dim3(HIDDEN / 128, MB), 256, 0, stream>>>(
            xb, lw3, g3, MROWS, HIDDEN, DMODEL, 0);
        const size_t nact = (size_t)MROWS * HIDDEN;
        k_silu_mul<<<(unsigned)((nact + 255) / 256), 256, 0, stream>>>(g1, g3, actb, nact);
        k_gemm_bf16<<<dim3(DMODEL / 128, MB), 256, 0, stream>>>(
            actb, lw2, h, MROWS, DMODEL, HIDDEN, 1);
    }

    k_rmsnorm_bf16<<<MROWS, 256, 0, stream>>>(h, final_w, xb);
    k_gemm_bf16<<<dim3(VOCAB / 128, MB), 256, 0, stream>>>(
        xb, headT, out, MROWS, VOCAB, DMODEL, 0);
}
